// IsoDecoder_66941360275959
// MI455X (gfx1250) — compile-verified, matrix operand order checked by reading, not run on hardware
//
#include <hip/hip_runtime.h>

#define D_MODEL 512
#define DHALF   256
#define SEQLEN  2048
#define BATCH   4
#define NEXP    4
#define NLAYER  4
#define NHEAD   8

typedef __attribute__((ext_vector_type(16))) __bf16 v16bf;
typedef __attribute__((ext_vector_type(8)))  float  v8f;

enum { MODE_U = 0, MODE_X2 = 1, MODE_FF1 = 2, MODE_FF2 = 3 };

// Async global->LDS staging path (CDNA5 GLOBAL_LOAD_ASYNC_TO_LDS_B128,
// ASYNCcnt-tracked). Builtin params are int4-vector pointers in AS1/AS3.
#if defined(__AMDGCN__) && \
    __has_builtin(__builtin_amdgcn_global_load_async_to_lds_b128) && \
    __has_builtin(__builtin_amdgcn_s_wait_asynccnt)
#define USE_ASYNC_LDS 1
typedef int vint4_t __attribute__((__vector_size__(4 * sizeof(int))));
typedef __attribute__((address_space(1))) vint4_t as1_int4;
typedef __attribute__((address_space(3))) vint4_t as3_int4;
#else
#define USE_ASYNC_LDS 0
#endif

__device__ __forceinline__ __bf16 f2bf(float f) {
  unsigned u; __builtin_memcpy(&u, &f, 4);
  unsigned r = u + 0x7FFFu + ((u >> 16) & 1u);   // round-to-nearest-even
  unsigned short h = (unsigned short)(r >> 16);
  __bf16 b; __builtin_memcpy(&b, &h, 2);
  return b;
}
__device__ __forceinline__ float bf2f(__bf16 b) {
  unsigned short s; __builtin_memcpy(&s, &b, 2);
  unsigned u = ((unsigned)s) << 16;
  float f; __builtin_memcpy(&f, &u, 4);
  return f;
}
__device__ __forceinline__ float elu1(float v) {
  return v > 0.f ? v : (__expf(v) - 1.f);
}

// ---------------------------------------------------------------------------
// WMMA GEMM:  C[m,n] = sum_k A[m,k] * W[n,k]   (A, W row-major bf16, K=D=512)
// Block tile 128x128, BK=32, 256 threads = 8 waves; wave tile 32x64
// (2x4 of 16x16), accumulate f32 via v_wmma_f32_16x16x32_bf16.
// Fused epilogues select per `mode`.
// ---------------------------------------------------------------------------
__global__ __launch_bounds__(256)
void wmma_gemm(int mode, int layer,
               const __bf16* __restrict__ A,
               const __bf16* __restrict__ W,
               const float*  __restrict__ bias,
               const float*  __restrict__ eigvec,     // MODE_U
               const float*  __restrict__ uxrow,      // MODE_X2
               const float*  __restrict__ xin,        // MODE_X2 (f32 x)
               const float*  __restrict__ rand_keys,  // MODE_X2
               const float*  __restrict__ resweight,  // MODE_X2 / MODE_FF2
               const float*  __restrict__ outc,       // MODE_FF2 residual
               __bf16*       __restrict__ obf,
               float*        __restrict__ of32)       // MODE_FF2 x out
{
  __shared__ __align__(16) __bf16 As[128 * 40];  // +8 halves pad
  __shared__ __align__(16) __bf16 Bs[128 * 40];

  const int tid = threadIdx.x;
  const int z   = blockIdx.z;          // slice = b*E+e (modes U/X2), else 0
  const int b   = z / NEXP;
  const int e   = z - b * NEXP;

  long long aoff, woff, boff, ooff;
  if (mode == MODE_U) {
    aoff = (long long)b * SEQLEN * D_MODEL;
    woff = (long long)z * D_MODEL * D_MODEL;
    boff = (long long)z * D_MODEL;
    ooff = (long long)z * SEQLEN * D_MODEL;
  } else if (mode == MODE_X2) {
    aoff = (long long)z * SEQLEN * D_MODEL;
    woff = (long long)z * D_MODEL * D_MODEL;
    boff = (long long)z * D_MODEL;
    ooff = (long long)z * SEQLEN * D_MODEL;
  } else {
    aoff = 0; woff = 0; boff = 0; ooff = 0;
  }
  const __bf16* Ab = A + aoff;
  const __bf16* Wb = W + woff;
  const float*  bb = bias + boff;

  const int m0 = blockIdx.y * 128;
  const int n0 = blockIdx.x * 128;

  // global -> LDS staging: each thread moves 32B (two b128 transfers);
  // the two threads of a row cover the full 32-half (64B) row fragment.
  const int lrow = tid >> 1;           // 0..127
  const int lcol = (tid & 1) * 16;     // 0 or 16 (halves)

  const int wid  = tid >> 5;
  const int lane = tid & 31;
  const int wm   = wid >> 1;           // 0..3 -> 32-row band
  const int wn   = wid & 1;            // 0..1 -> 64-col band
  const int lr   = lane & 15;
  const int hi8  = (lane >> 4) & 1;    // upper half-wave
  const int kb   = hi8 * 8;            // K base per ISA 16-bit A layout

  v8f acc[2][4];
  v8f zero = {0.f, 0.f, 0.f, 0.f, 0.f, 0.f, 0.f, 0.f};
#pragma unroll
  for (int i = 0; i < 2; ++i)
#pragma unroll
    for (int j = 0; j < 4; ++j) acc[i][j] = zero;

  for (int k0 = 0; k0 < D_MODEL; k0 += 32) {
    const __bf16* ga = Ab + (long long)(m0 + lrow) * D_MODEL + k0 + lcol;
    const __bf16* gw = Wb + (long long)(n0 + lrow) * D_MODEL + k0 + lcol;
#if USE_ASYNC_LDS
    __builtin_amdgcn_global_load_async_to_lds_b128(
        (as1_int4*)ga,       (as3_int4*)&As[lrow * 40 + lcol], 0, 0);
    __builtin_amdgcn_global_load_async_to_lds_b128(
        (as1_int4*)(ga + 8), (as3_int4*)&As[lrow * 40 + lcol + 8], 0, 0);
    __builtin_amdgcn_global_load_async_to_lds_b128(
        (as1_int4*)gw,       (as3_int4*)&Bs[lrow * 40 + lcol], 0, 0);
    __builtin_amdgcn_global_load_async_to_lds_b128(
        (as1_int4*)(gw + 8), (as3_int4*)&Bs[lrow * 40 + lcol + 8], 0, 0);
#else
    *(uint4*)&As[lrow * 40 + lcol]     = *(const uint4*)ga;
    *(uint4*)&As[lrow * 40 + lcol + 8] = *(const uint4*)(ga + 8);
    *(uint4*)&Bs[lrow * 40 + lcol]     = *(const uint4*)gw;
    *(uint4*)&Bs[lrow * 40 + lcol + 8] = *(const uint4*)(gw + 8);
#endif
    if (k0 + 32 < D_MODEL) {           // global_prefetch_b8 of next K slab
      __builtin_prefetch(ga + 32, 0, 1);
      __builtin_prefetch(gw + 32, 0, 1);
    }
#if USE_ASYNC_LDS
    __builtin_amdgcn_s_wait_asynccnt(0);
#endif
    __syncthreads();

    union Frag { uint4 u[2]; v16bf v; } afr[2], bfr[4];
#pragma unroll
    for (int tm = 0; tm < 2; ++tm) {
      int row = wm * 32 + tm * 16 + lr;
      afr[tm].u[0] = *(const uint4*)&As[row * 40 + kb];
      afr[tm].u[1] = *(const uint4*)&As[row * 40 + kb + 16];
    }
#pragma unroll
    for (int tn = 0; tn < 4; ++tn) {
      int col = wn * 64 + tn * 16 + lr;
      bfr[tn].u[0] = *(const uint4*)&Bs[col * 40 + kb];
      bfr[tn].u[1] = *(const uint4*)&Bs[col * 40 + kb + 16];
    }
#pragma unroll
    for (int tm = 0; tm < 2; ++tm)
#pragma unroll
      for (int tn = 0; tn < 4; ++tn)
        acc[tm][tn] = __builtin_amdgcn_wmma_f32_16x16x32_bf16(
            false, afr[tm].v, false, bfr[tn].v,
            (short)0, acc[tm][tn], false, false);
    __syncthreads();
  }

  float rw = 0.f;
  if (mode == MODE_X2 || mode == MODE_FF2)
    rw = tanhf(resweight[layer]) * (1.0f / NLAYER);

#pragma unroll
  for (int tm = 0; tm < 2; ++tm) {
#pragma unroll
    for (int tn = 0; tn < 4; ++tn) {
#pragma unroll
      for (int r = 0; r < 8; ++r) {
        int m = m0 + wm * 32 + tm * 16 + r + hi8 * 8;
        int n = n0 + wn * 64 + tn * 16 + lr;
        float v = acc[tm][tn][r];
        if (mode == MODE_U) {
          // Uxs = (x@Uw^T + Ub) * e_b[n]
          float o = (v + bb[n]) * eigvec[b * D_MODEL + n];
          obf[ooff + (long long)m * D_MODEL + n] = f2bf(o);
        } else if (mode == MODE_X2) {
          // x2 = elu(Uxs@Vw^T + Vb*rowsum)*rw ; out = a*x + c*x2
          float t  = v + bb[n] * uxrow[(long long)z * SEQLEN + m];
          float x2 = elu1(t) * rw;
          float rk = rand_keys[(layer * NEXP + e) * NHEAD + (n & 7)];
          float av = (rk <= 0.25f) ? 1.5f : (rk <= 0.5f) ? 1.5f
                    : (rk <= 0.75f) ? 0.f : 1.f;
          float cv = (rk <= 0.25f) ? 1.f : (rk <= 0.5f) ? -1.f
                    : (rk <= 0.75f) ? 4.f : 0.f;
          float xv = xin[((long long)b * SEQLEN + m) * D_MODEL + n];
          obf[ooff + (long long)m * D_MODEL + n] = f2bf(av * xv + cv * x2);
        } else if (mode == MODE_FF1) {
          obf[(long long)m * D_MODEL + n] = f2bf(elu1(v + bb[n]));
        } else { // MODE_FF2: x_new = out + rw*elu(h1@W^T + b)
          float h2 = elu1(v + bb[n]);
          float xn = outc[(long long)m * D_MODEL + n] + rw * h2;
          of32[(long long)m * D_MODEL + n] = xn;
          obf [(long long)m * D_MODEL + n] = f2bf(xn);
        }
      }
    }
  }
}

// ---------------------------------------------------------------------------
// Support kernels
// ---------------------------------------------------------------------------
__global__ void k_convert(const float* __restrict__ src,
                          __bf16* __restrict__ dst, int n) {
  int i = blockIdx.x * blockDim.x + threadIdx.x;
  if (i < n) dst[i] = f2bf(src[i]);
}

__global__ void k_embed(const int* __restrict__ src,
                        const float* __restrict__ emb,
                        const float* __restrict__ pe,
                        float* __restrict__ x, __bf16* __restrict__ xb) {
  int idx = blockIdx.x * blockDim.x + threadIdx.x;   // B*S*D exact
  int d = idx % D_MODEL;
  int t = idx / D_MODEL;                              // b*S + s
  int s = t % SEQLEN;
  float v = (d < DHALF) ? emb[(long long)src[t] * DHALF + d]
                        : pe[s * DHALF + (d - DHALF)];
  x[idx]  = v;
  xb[idx] = f2bf(v);
}

__global__ void k_eig(const float* __restrict__ eig_raw,
                      float* __restrict__ eigvec) {
  __shared__ float red[512];
  int b = blockIdx.x, t = threadIdx.x;                // 512 threads
  float v = tanhf(eig_raw[b * D_MODEL + t]);
  red[t] = fabsf(v);
  __syncthreads();
  for (int s2 = 256; s2 > 0; s2 >>= 1) {
    if (t < s2) red[t] = fmaxf(red[t], red[t + s2]);
    __syncthreads();
  }
  eigvec[b * D_MODEL + t] = v / red[0];
}

// colsum[i,e,k] = sum_d U_w[i,e,d,k]
__global__ void k_colsum(const float* __restrict__ Uw,
                         float* __restrict__ colsum) {
  int idx = blockIdx.x * blockDim.x + threadIdx.x;
  if (idx >= NLAYER * NEXP * D_MODEL) return;
  int k = idx % D_MODEL;
  int ie = idx / D_MODEL;
  const float* base = Uw + (long long)ie * D_MODEL * D_MODEL;
  float s = 0.f;
  for (int d0 = 0; d0 < D_MODEL; ++d0)
    s += base[(long long)d0 * D_MODEL + k];
  colsum[idx] = s;
}

// uxrow[b,e,s] = x[b,s,:]·colsum[i,e,:] + sum(U_b[i,e,:])
__global__ void k_rowsum(int layer, const float* __restrict__ x,
                         const float* __restrict__ colsum,
                         const float* __restrict__ Ub,
                         float* __restrict__ uxrow) {
  int idx = blockIdx.x * blockDim.x + threadIdx.x;    // B*E*S
  if (idx >= BATCH * NEXP * SEQLEN) return;
  int s = idx % SEQLEN;
  int z = idx / SEQLEN;
  int b = z / NEXP, e = z % NEXP;
  const float* xr = x + ((long long)b * SEQLEN + s) * D_MODEL;
  const float* cs = colsum + (long long)(layer * NEXP + e) * D_MODEL;
  const float* ub = Ub + (long long)(layer * NEXP + e) * D_MODEL;
  float acc = 0.f, sb = 0.f;
#pragma unroll 4
  for (int k = 0; k < D_MODEL; ++k) { acc += xr[k] * cs[k]; sb += ub[k]; }
  uxrow[idx] = acc + sb;
}

// logits[b,e] = (max_s out_e[b,e,s,:])·comb_w + comb_b
__global__ void k_logits(int layer, const __bf16* __restrict__ oute,
                         const float* __restrict__ combw,
                         const float* __restrict__ combb,
                         float* __restrict__ logits) {
  __shared__ float red[256];
  int z = blockIdx.x;                                 // b*E+e
  int t = threadIdx.x;                                // 256
  float m1 = -3.4e38f, m2 = -3.4e38f;
  const __bf16* base = oute + (long long)z * SEQLEN * D_MODEL;
  for (int s = 0; s < SEQLEN; ++s) {
    m1 = fmaxf(m1, bf2f(base[(long long)s * D_MODEL + t]));
    m2 = fmaxf(m2, bf2f(base[(long long)s * D_MODEL + t + 256]));
  }
  red[t] = m1 * combw[layer * D_MODEL + t] + m2 * combw[layer * D_MODEL + t + 256];
  __syncthreads();
  for (int s2 = 128; s2 > 0; s2 >>= 1) {
    if (t < s2) red[t] += red[t + s2];
    __syncthreads();
  }
  if (t == 0) logits[z] = red[0] + combb[layer];
}

__global__ void k_softmax(const float* __restrict__ logits,
                          float* __restrict__ wv) {
  int b = threadIdx.x;
  if (b < BATCH) {
    float mx = -3.4e38f;
    for (int e = 0; e < NEXP; ++e) mx = fmaxf(mx, logits[b * NEXP + e]);
    float s = 0.f, ex[NEXP];
    for (int e = 0; e < NEXP; ++e) { ex[e] = __expf(logits[b * NEXP + e] - mx); s += ex[e]; }
    for (int e = 0; e < NEXP; ++e) wv[b * NEXP + e] = ex[e] / s;
  }
}

__global__ void k_combine(const __bf16* __restrict__ oute,
                          const float* __restrict__ wv,
                          float* __restrict__ outc,
                          __bf16* __restrict__ outcb) {
  int idx = blockIdx.x * blockDim.x + threadIdx.x;    // B*S*D exact
  int b = idx / (SEQLEN * D_MODEL);
  int rem = idx - b * SEQLEN * D_MODEL;
  float acc = 0.f;
#pragma unroll
  for (int e = 0; e < NEXP; ++e)
    acc += wv[b * NEXP + e] *
           bf2f(oute[(long long)(b * NEXP + e) * SEQLEN * D_MODEL + rem]);
  outc[idx]  = acc;
  outcb[idx] = f2bf(acc);
}

__global__ void k_copyout(const float* __restrict__ x, float* __restrict__ out) {
  int idx = blockIdx.x * blockDim.x + threadIdx.x;
  if (idx < BATCH * SEQLEN * D_MODEL) out[idx] = x[idx];
  if (idx == 0) out[BATCH * SEQLEN * D_MODEL] = 0.f;  // recon_loss
}

// ---------------------------------------------------------------------------
extern "C" void kernel_launch(void* const* d_in, const int* in_sizes, int n_in,
                              void* d_out, int out_size, void* d_ws, size_t ws_size,
                              hipStream_t stream) {
  const int*   src   = (const int*)d_in[0];
  const float* emb   = (const float*)d_in[1];
  const float* pe    = (const float*)d_in[2];
  const float* Uw    = (const float*)d_in[3];
  const float* Ub    = (const float*)d_in[4];
  const float* Vw    = (const float*)d_in[5];
  const float* Vb    = (const float*)d_in[6];
  const float* combw = (const float*)d_in[7];
  const float* combb = (const float*)d_in[8];
  const float* ff1w  = (const float*)d_in[9];
  const float* ff1bp = (const float*)d_in[10];
  const float* ff2w  = (const float*)d_in[11];
  const float* ff2bp = (const float*)d_in[12];
  const float* resw  = (const float*)d_in[13];
  const float* eigr  = (const float*)d_in[14];
  const float* rkeys = (const float*)d_in[15];
  (void)in_sizes; (void)n_in; (void)out_size; (void)ws_size;

  char* p = (char*)d_ws;
  auto alloc = [&](size_t bytes) -> void* {
    void* r = p; p += (bytes + 255) & ~(size_t)255; return r;
  };

  const long long XSZ = (long long)BATCH * SEQLEN * D_MODEL;     // 4M
  float*  x_f32 = (float*) alloc(XSZ * 4);
  __bf16* x_bf  = (__bf16*)alloc(XSZ * 2);
  __bf16* Uwb   = (__bf16*)alloc((long long)NLAYER * NEXP * D_MODEL * D_MODEL * 2);
  __bf16* Vwb   = (__bf16*)alloc((long long)NLAYER * NEXP * D_MODEL * D_MODEL * 2);
  __bf16* f1wb  = (__bf16*)alloc((long long)NLAYER * D_MODEL * D_MODEL * 2);
  __bf16* f2wb  = (__bf16*)alloc((long long)NLAYER * D_MODEL * D_MODEL * 2);
  __bf16* Uxs   = (__bf16*)alloc((long long)NEXP * XSZ * 2);     // [B,E,S,D]
  __bf16* oute  = (__bf16*)alloc((long long)NEXP * XSZ * 2);     // [B,E,S,D]
  float*  outc  = (float*) alloc(XSZ * 4);
  __bf16* outcb = (__bf16*)alloc(XSZ * 2);
  __bf16* h1b   = (__bf16*)alloc(XSZ * 2);
  float*  uxrow = (float*) alloc((long long)BATCH * NEXP * SEQLEN * 4);
  float*  csum  = (float*) alloc((long long)NLAYER * NEXP * D_MODEL * 4);
  float*  eigv  = (float*) alloc((long long)BATCH * D_MODEL * 4);
  float*  logit = (float*) alloc(BATCH * NEXP * 4);
  float*  wvec  = (float*) alloc(BATCH * NEXP * 4);

  const int nUw = NLAYER * NEXP * D_MODEL * D_MODEL;   // 4,194,304
  const int nFw = NLAYER * D_MODEL * D_MODEL;          // 1,048,576
  k_convert<<<(nUw + 255) / 256, 256, 0, stream>>>(Uw,   Uwb,  nUw);
  k_convert<<<(nUw + 255) / 256, 256, 0, stream>>>(Vw,   Vwb,  nUw);
  k_convert<<<(nFw + 255) / 256, 256, 0, stream>>>(ff1w, f1wb, nFw);
  k_convert<<<(nFw + 255) / 256, 256, 0, stream>>>(ff2w, f2wb, nFw);
  k_colsum<<<(NLAYER * NEXP * D_MODEL + 255) / 256, 256, 0, stream>>>(Uw, csum);
  k_embed<<<(int)(XSZ / 256), 256, 0, stream>>>(src, emb, pe, x_f32, x_bf);
  k_eig<<<BATCH, 512, 0, stream>>>(eigr, eigv);

  dim3 gU(D_MODEL / 128, SEQLEN / 128, BATCH * NEXP);          // 4,16,16
  dim3 gF(D_MODEL / 128, (BATCH * SEQLEN) / 128, 1);           // 4,64,1

  for (int L = 0; L < NLAYER; ++L) {
    k_rowsum<<<(BATCH * NEXP * SEQLEN + 255) / 256, 256, 0, stream>>>(
        L, x_f32, csum, Ub, uxrow);

    wmma_gemm<<<gU, 256, 0, stream>>>(MODE_U, L,
        x_bf, Uwb + (long long)L * NEXP * D_MODEL * D_MODEL,
        Ub + L * NEXP * D_MODEL,
        eigv, nullptr, nullptr, nullptr, nullptr, nullptr,
        Uxs, nullptr);

    wmma_gemm<<<gU, 256, 0, stream>>>(MODE_X2, L,
        Uxs, Vwb + (long long)L * NEXP * D_MODEL * D_MODEL,
        Vb + L * NEXP * D_MODEL,
        nullptr, uxrow, x_f32, rkeys, resw, nullptr,
        oute, nullptr);

    k_logits<<<BATCH * NEXP, 256, 0, stream>>>(L, oute, combw, combb, logit);
    k_softmax<<<1, 32, 0, stream>>>(logit, wvec);
    k_combine<<<(int)(XSZ / 256), 256, 0, stream>>>(oute, wvec, outc, outcb);

    wmma_gemm<<<gF, 256, 0, stream>>>(MODE_FF1, L,
        outcb, f1wb + (long long)L * D_MODEL * D_MODEL, ff1bp + L * D_MODEL,
        nullptr, nullptr, nullptr, nullptr, nullptr, nullptr,
        h1b, nullptr);

    wmma_gemm<<<gF, 256, 0, stream>>>(MODE_FF2, L,
        h1b, f2wb + (long long)L * D_MODEL * D_MODEL, ff2bp + L * D_MODEL,
        nullptr, nullptr, nullptr, nullptr, resw, outc,
        x_bf, x_f32);
  }

  k_copyout<<<(int)((XSZ + 256) / 256), 256, 0, stream>>>(x_f32, (float*)d_out);
}
